// EdgeSparkNet_26061861552524
// MI455X (gfx1250) — matrix operand order repaired and checked by reading, not compiled, not run on hardware
//
#include <hip/hip_runtime.h>

// ---------------------------------------------------------------------------
// Problem constants (from reference)
// ---------------------------------------------------------------------------
#define BB 128
#define NN 50
#define DD 256
#define MN (BB * NN)      // 6400 rows
#define FP 64
#define SS (NN * NN)      // 2500
#define LN_EPS 1e-5f

typedef __attribute__((ext_vector_type(16))) __bf16 v16bf;
typedef __attribute__((ext_vector_type(8)))  float  v8f;

// ---------------------------------------------------------------------------
// Operand packing: fp32 row-major -> bf16 in WMMA fragment order (wave32,
// ISA 7.12.2). Layout: [tile][lane 0..31][elem 0..15], 32B per lane.
//   A (16x32 MxK per tile): tile = mtile*KT + ktile
//     lane = g*16 + (m%16);  elem e=2j+h -> k = ktile*32 + (j&4?16:0)+g*8+(j&3)*2+h
//   B (32x16 KxN per tile): tile = ktile*NT + ntile
//     lane = (k'&8)*2 + (n%16); elem e = (k'&16)/2 + (k'&7)   (k' = k - ktile*32)
// ---------------------------------------------------------------------------
__global__ __launch_bounds__(256) void pack_a_kernel(
    const float* __restrict__ X, __bf16* __restrict__ P, int M, int K)
{
    int o = blockIdx.x * 256 + threadIdx.x;
    if (o >= M * K) return;
    int e    = o & 15;
    int lane = (o >> 4) & 31;
    int tile = o >> 9;
    int KT   = K >> 5;
    int mtile = tile / KT, ktile = tile % KT;
    int g = lane >> 4, lm = lane & 15;
    int j = e >> 1, h = e & 1;
    int k = ktile * 32 + ((j & 4) ? 16 : 0) + g * 8 + (j & 3) * 2 + h;
    int m = mtile * 16 + lm;
    P[o] = (__bf16)X[(size_t)m * K + k];
}

__global__ __launch_bounds__(256) void pack_b_kernel(
    const float* __restrict__ W, __bf16* __restrict__ P, int K, int N)
{
    int o = blockIdx.x * 256 + threadIdx.x;
    if (o >= K * N) return;
    int e    = o & 15;
    int lane = (o >> 4) & 31;
    int tile = o >> 9;
    int NT   = N >> 4;
    int ktile = tile / NT, ntile = tile % NT;
    int k = ktile * 32 + ((e & 8) ? 16 : 0) + (lane >> 4) * 8 + (e & 7);
    int n = ntile * 16 + (lane & 15);
    P[o] = (__bf16)W[(size_t)k * N + n];
}

// ---------------------------------------------------------------------------
// WMMA bf16 GEMM on pre-packed operands:  C[M,N] = unpack(Ap) @ unpack(Bp)
// (+ bias, optional ReLU). No LDS, no barriers: each lane's fragment is two
// coalesced global_load_b128 (entire working set resides in L2).
// Block = 128 threads (4 waves); wave w -> mtile blockIdx.x*4+w, 4 N-subtiles.
// Requires M%64==0, N%64==0, K%32==0 (true for all call sites).
// ---------------------------------------------------------------------------
template <int RELU, int HASBIAS>
__global__ __launch_bounds__(128) void gemm_wmma_packed(
    const __bf16* __restrict__ Ap, const __bf16* __restrict__ Bp,
    const float* __restrict__ bias, float* __restrict__ C,
    int M, int N, int K)
{
    const int wave = threadIdx.x >> 5;
    const int lane = threadIdx.x & 31;
    const int g    = lane >> 4;
    const int lm   = lane & 15;

    const int mtile = blockIdx.x * 4 + wave;
    const int nt0   = blockIdx.y * 4;
    const int KT    = K >> 5;
    const int NT    = N >> 4;

    const v16bf* Af = (const v16bf*)Ap + (size_t)mtile * KT * 32 + lane;
    const v16bf* Bf = (const v16bf*)Bp + lane;

    v8f acc0 = {}, acc1 = {}, acc2 = {}, acc3 = {};

    for (int kt = 0; kt < KT; ++kt) {
        const v16bf* B0 = Bf + ((size_t)kt * NT + nt0) * 32;
        if (kt + 1 < KT) {
            // prefetch next K-step's B fragments (global_prefetch_b8)
            __builtin_prefetch((const void*)(Bf + ((size_t)(kt + 1) * NT + nt0) * 32), 0, 1);
        }
        v16bf a  = Af[(size_t)kt * 32];
        v16bf b0 = B0[0];
        v16bf b1 = B0[32];
        v16bf b2 = B0[64];
        v16bf b3 = B0[96];

        acc0 = __builtin_amdgcn_wmma_f32_16x16x32_bf16(false, a, false, b0, (short)0, acc0, false, false);
        acc1 = __builtin_amdgcn_wmma_f32_16x16x32_bf16(false, a, false, b1, (short)0, acc1, false, false);
        acc2 = __builtin_amdgcn_wmma_f32_16x16x32_bf16(false, a, false, b2, (short)0, acc2, false, false);
        acc3 = __builtin_amdgcn_wmma_f32_16x16x32_bf16(false, a, false, b3, (short)0, acc3, false, false);
    }

    // Epilogue: C/D layout -> row m = g*8 + j, col n = lane&15
    v8f* accs[4] = { &acc0, &acc1, &acc2, &acc3 };
    #pragma unroll
    for (int nt = 0; nt < 4; ++nt) {
        int col = (nt0 + nt) * 16 + lm;
        float bcol = HASBIAS ? bias[col] : 0.0f;
        v8f av = *accs[nt];
        #pragma unroll
        for (int j = 0; j < 8; ++j) {
            int row = mtile * 16 + g * 8 + j;
            float v = av[j] + bcol;
            if (RELU) v = fmaxf(v, 0.0f);
            C[(size_t)row * N + col] = v;
        }
    }
}

// ---------------------------------------------------------------------------
// Per-token 8x8 head attention (contract over hd=32 within one token).
// One thread per (token, head).
// ---------------------------------------------------------------------------
__global__ __launch_bounds__(256) void head_attn_kernel(
    const float* __restrict__ q, const float* __restrict__ k,
    const float* __restrict__ v, float* __restrict__ att, int tokens)
{
    int t = blockIdx.x * blockDim.x + threadIdx.x;
    if (t >= tokens * 8) return;
    int tok = t >> 3, h = t & 7;
    const float* qr = q + (size_t)tok * DD + h * 32;
    const float* kb = k + (size_t)tok * DD;
    const float* vb = v + (size_t)tok * DD;

    float qv[32];
    #pragma unroll
    for (int d = 0; d < 32; ++d) qv[d] = qr[d];

    const float inv = 0.17677669529663687f;  // 1/sqrt(32)
    float sc[8], mx = -1e30f;
    for (int gg = 0; gg < 8; ++gg) {
        float s = 0.f;
        #pragma unroll
        for (int d = 0; d < 32; ++d) s += qv[d] * kb[gg * 32 + d];
        s *= inv;
        sc[gg] = s;
        mx = fmaxf(mx, s);
    }
    float sum = 0.f;
    for (int gg = 0; gg < 8; ++gg) { sc[gg] = __expf(sc[gg] - mx); sum += sc[gg]; }
    float isum = 1.0f / sum;

    float o[32];
    #pragma unroll
    for (int d = 0; d < 32; ++d) o[d] = 0.f;
    for (int gg = 0; gg < 8; ++gg) {
        float w = sc[gg] * isum;
        #pragma unroll
        for (int d = 0; d < 32; ++d) o[d] += w * vb[gg * 32 + d];
    }
    float* ar = att + (size_t)tok * DD + h * 32;
    #pragma unroll
    for (int d = 0; d < 32; ++d) ar[d] = o[d];
}

// ---------------------------------------------------------------------------
// out[row,:] = LayerNorm(resid[row,:] + y[row,:]) * gamma + beta, width 256
// ---------------------------------------------------------------------------
__global__ __launch_bounds__(256) void resln_kernel(
    const float* __restrict__ resid, const float* __restrict__ y,
    const float* __restrict__ gamma, const float* __restrict__ beta,
    float* __restrict__ out)
{
    int row = blockIdx.x, t = threadIdx.x;
    float x = resid[(size_t)row * DD + t] + y[(size_t)row * DD + t];
    __shared__ float red[256];
    red[t] = x; __syncthreads();
    for (int s = 128; s > 0; s >>= 1) { if (t < s) red[t] += red[t + s]; __syncthreads(); }
    float mu = red[0] * (1.0f / DD);
    __syncthreads();
    float dx = x - mu;
    red[t] = dx * dx; __syncthreads();
    for (int s = 128; s > 0; s >>= 1) { if (t < s) red[t] += red[t + s]; __syncthreads(); }
    float var = red[0] * (1.0f / DD);
    out[(size_t)row * DD + t] = dx * rsqrtf(var + LN_EPS) * gamma[t] + beta[t];
}

// ---------------------------------------------------------------------------
// Msym = 0.5 * (M + M^T), 256x256
// ---------------------------------------------------------------------------
__global__ void msym_kernel(const float* __restrict__ Min, float* __restrict__ Mout)
{
    int idx = blockIdx.x * blockDim.x + threadIdx.x;
    int i = idx >> 8, j = idx & 255;
    Mout[idx] = 0.5f * (Min[i * DD + j] + Min[j * DD + i]);
}

// ---------------------------------------------------------------------------
// qout[row] = sum_d tM[row,d] * t[row,d]
// ---------------------------------------------------------------------------
__global__ void rowq_kernel(const float* __restrict__ tM, const float* __restrict__ t,
                            float* __restrict__ qout, int rows)
{
    int r = blockIdx.x * blockDim.x + threadIdx.x;
    if (r >= rows) return;
    const float* a = tM + (size_t)r * DD;
    const float* b = t + (size_t)r * DD;
    float s = 0.f;
    for (int d = 0; d < DD; ++d) s += a[d] * b[d];
    qout[r] = s;
}

// ---------------------------------------------------------------------------
// Fused cross / dist / exp-sim : flat[b, n*50+m]
// ---------------------------------------------------------------------------
__global__ __launch_bounds__(256) void cross_sim_kernel(
    const float* __restrict__ t1M, const float* __restrict__ t2,
    const float* __restrict__ q1, const float* __restrict__ q2,
    float* __restrict__ flat)
{
    int b = blockIdx.x;
    int idx = blockIdx.y * 256 + threadIdx.x;
    if (idx >= SS) return;
    int n = idx / NN, m = idx % NN;
    const float* x = t1M + ((size_t)b * NN + n) * DD;
    const float* y = t2  + ((size_t)b * NN + m) * DD;
    float s = 0.f;
    for (int d = 0; d < DD; ++d) s += x[d] * y[d];
    float dist = q1[b * NN + n] + q2[b * NN + m] - 2.0f * s;
    float z = fminf(fmaxf(-dist, -10.0f), 10.0f);
    flat[(size_t)b * SS + idx] = __expf(z);
}

__global__ void zero_kernel(float* p, int n)
{
    int i = blockIdx.x * blockDim.x + threadIdx.x;
    if (i < n) p[i] = 0.f;
}

// ---------------------------------------------------------------------------
// Attention pooling: per (b,s) an 8x8 head attention over hdp=8; accumulate
// sum over s into accum[b, 64]. Block = 256 threads = 32 tokens x 8 heads.
// ---------------------------------------------------------------------------
__global__ __launch_bounds__(256) void pool_attn_kernel(
    const float* __restrict__ flat,
    const float* __restrict__ wq, const float* __restrict__ bq,
    const float* __restrict__ wk, const float* __restrict__ bk,
    const float* __restrict__ wv, const float* __restrict__ bv,
    const float* __restrict__ pos, float* __restrict__ accum)
{
    __shared__ float acc[FP];
    int t = threadIdx.x;
    if (t < FP) acc[t] = 0.f;
    __syncthreads();

    int b = blockIdx.x;
    int s = blockIdx.y * 32 + (t >> 3);
    int h = t & 7;
    if (s < SS) {
        float f = flat[(size_t)b * SS + s];
        const float* ps = pos + (size_t)s * FP;
        float qh[8];
        #pragma unroll
        for (int j = 0; j < 8; ++j)
            qh[j] = f * wq[h * 8 + j] + bq[h * 8 + j] + ps[h * 8 + j];

        const float inv = 0.35355339059327373f;  // 1/sqrt(8)
        float sc[8], mx = -1e30f;
        for (int g = 0; g < 8; ++g) {
            float d = 0.f;
            #pragma unroll
            for (int j = 0; j < 8; ++j) {
                float kg = f * wk[g * 8 + j] + bk[g * 8 + j] + ps[g * 8 + j];
                d += qh[j] * kg;
            }
            d *= inv;
            sc[g] = d;
            mx = fmaxf(mx, d);
        }
        float sum = 0.f;
        for (int g = 0; g < 8; ++g) { sc[g] = __expf(sc[g] - mx); sum += sc[g]; }
        float is = 1.0f / sum;

        float o[8];
        #pragma unroll
        for (int j = 0; j < 8; ++j) o[j] = 0.f;
        for (int g = 0; g < 8; ++g) {
            float w = sc[g] * is;
            #pragma unroll
            for (int j = 0; j < 8; ++j)
                o[j] += w * (f * wv[g * 8 + j] + bv[g * 8 + j]);
        }
        #pragma unroll
        for (int j = 0; j < 8; ++j) atomicAdd(&acc[h * 8 + j], o[j]);
    }
    __syncthreads();
    if (t < FP) atomicAdd(&accum[(size_t)b * FP + t], acc[t]);
}

// ---------------------------------------------------------------------------
// Final: out[b,:] = LN( (accum[b,:]/S) @ wo + bo ), width 64. Block per b.
// ---------------------------------------------------------------------------
__global__ __launch_bounds__(64) void pool_final_kernel(
    const float* __restrict__ accum, const float* __restrict__ wo,
    const float* __restrict__ bo, const float* __restrict__ gamma,
    const float* __restrict__ beta, float* __restrict__ out)
{
    int b = blockIdx.x, t = threadIdx.x;
    __shared__ float p[FP];
    __shared__ float red[FP];
    p[t] = accum[(size_t)b * FP + t] * (1.0f / (float)SS);
    __syncthreads();
    float y = bo[t];
    for (int j = 0; j < FP; ++j) y += p[j] * wo[j * FP + t];
    red[t] = y; __syncthreads();
    for (int s = 32; s > 0; s >>= 1) { if (t < s) red[t] += red[t + s]; __syncthreads(); }
    float mu = red[0] * (1.0f / FP);
    __syncthreads();
    float dy = y - mu;
    red[t] = dy * dy; __syncthreads();
    for (int s = 32; s > 0; s >>= 1) { if (t < s) red[t] += red[t + s]; __syncthreads(); }
    float var = red[0] * (1.0f / FP);
    out[(size_t)b * FP + t] = dy * rsqrtf(var + LN_EPS) * gamma[t] + beta[t];
}

// ---------------------------------------------------------------------------
// Host orchestration
// ---------------------------------------------------------------------------
extern "C" void kernel_launch(void* const* d_in, const int* in_sizes, int n_in,
                              void* d_out, int out_size, void* d_ws, size_t ws_size,
                              hipStream_t stream)
{
    const float* f1      = (const float*)d_in[0];
    const float* f2      = (const float*)d_in[1];
    const float* ca_wq   = (const float*)d_in[2];
    const float* ca_bq   = (const float*)d_in[3];
    const float* ca_wk   = (const float*)d_in[4];
    const float* ca_bk   = (const float*)d_in[5];
    const float* ca_wv   = (const float*)d_in[6];
    const float* ca_bv   = (const float*)d_in[7];
    const float* ca_wo   = (const float*)d_in[8];
    const float* ca_bo   = (const float*)d_in[9];
    const float* ca_ln1g = (const float*)d_in[10];
    const float* ca_ln1b = (const float*)d_in[11];
    const float* ca_ln2g = (const float*)d_in[12];
    const float* ca_ln2b = (const float*)d_in[13];
    const float* ca_w1   = (const float*)d_in[14];
    const float* ca_b1   = (const float*)d_in[15];
    const float* ca_w2   = (const float*)d_in[16];
    const float* ca_b2   = (const float*)d_in[17];
    const float* ms_met  = (const float*)d_in[18];
    const float* ms_wt   = (const float*)d_in[19];
    const float* ms_bt   = (const float*)d_in[20];
    const float* ap_wq   = (const float*)d_in[21];
    const float* ap_bq   = (const float*)d_in[22];
    const float* ap_wk   = (const float*)d_in[23];
    const float* ap_bk   = (const float*)d_in[24];
    const float* ap_wv   = (const float*)d_in[25];
    const float* ap_bv   = (const float*)d_in[26];
    const float* ap_pos  = (const float*)d_in[27];
    const float* ap_wo   = (const float*)d_in[28];
    const float* ap_bo   = (const float*)d_in[29];
    const float* ap_lng  = (const float*)d_in[30];
    const float* ap_lnb  = (const float*)d_in[31];

    // ---------------- workspace layout ----------------
    float* ws = (float*)d_ws;
    const size_t SZ = (size_t)MN * DD;        // 1,638,400 floats
    float* s0   = ws;                          // q / x / t1
    float* s1   = s0 + SZ;                     // k / tmp / h2 / t2
    float* s2   = s1 + SZ;                     // v / t1M
    float* s3   = s2 + SZ;                     // att / t2M
    float* sH   = s3 + SZ;                     // FFN hidden (6400 x 512)
    float* e1   = sH + (size_t)MN * 2 * DD;
    float* e2   = e1 + SZ;
    float* flat = e2 + SZ;                     // 128 x 2500
    float* msym = flat + (size_t)BB * SS;      // 256 x 256
    float* q1w  = msym + (size_t)DD * DD;      // 6400
    float* q2w  = q1w + MN;                    // 6400
    float* accu = q2w + MN;                    // 128 x 64

    // bf16 packed operand pool (after fp32 region; offsets in bf16 elems)
    __bf16* bp = (__bf16*)(accu + (size_t)BB * FP);
    const size_t WSZ = (size_t)DD * DD;        // 65,536
    __bf16* wq_p   = bp;                       // 256x256
    __bf16* wk_p   = wq_p + WSZ;
    __bf16* wv_p   = wk_p + WSZ;
    __bf16* wo_p   = wv_p + WSZ;
    __bf16* mswt_p = wo_p + WSZ;
    __bf16* msym_p = mswt_p + WSZ;
    __bf16* w1_p   = msym_p + WSZ;             // 256x512
    __bf16* w2_p   = w1_p + 2 * WSZ;           // 512x256
    __bf16* pA     = w2_p + 2 * WSZ;           // 6400x256 activations
    __bf16* pB     = pA + SZ;                  // 6400x256
    __bf16* pH     = pB + SZ;                  // 6400x512

    auto packA = [&](const float* X, __bf16* P, int M, int K) {
        pack_a_kernel<<<(M * K) / 256, 256, 0, stream>>>(X, P, M, K);
    };
    auto packB = [&](const float* W, __bf16* P, int K, int N) {
        pack_b_kernel<<<(K * N) / 256, 256, 0, stream>>>(W, P, K, N);
    };
    auto gemm = [&](const __bf16* Ap, const __bf16* Bp2, const float* b, float* C,
                    int M, int N, int K, bool relu) {
        dim3 grid(M / 64, N / 64);
        if (relu)       gemm_wmma_packed<1, 1><<<grid, 128, 0, stream>>>(Ap, Bp2, b, C, M, N, K);
        else if (b)     gemm_wmma_packed<0, 1><<<grid, 128, 0, stream>>>(Ap, Bp2, b, C, M, N, K);
        else            gemm_wmma_packed<0, 0><<<grid, 128, 0, stream>>>(Ap, Bp2, b, C, M, N, K);
    };

    // ---- pack all static weights once
    packB(ca_wq, wq_p, DD, DD);
    packB(ca_wk, wk_p, DD, DD);
    packB(ca_wv, wv_p, DD, DD);
    packB(ca_wo, wo_p, DD, DD);
    packB(ms_wt, mswt_p, DD, DD);
    packB(ca_w1, w1_p, DD, 2 * DD);
    packB(ca_w2, w2_p, 2 * DD, DD);

    auto attend = [&](const float* fx, const float* fy, float* e) {
        packA(fx, pA, MN, DD);
        packA(fy, pB, MN, DD);
        gemm(pA, wq_p, ca_bq, s0, MN, DD, DD, false);           // Q
        gemm(pB, wk_p, ca_bk, s1, MN, DD, DD, false);           // K
        gemm(pB, wv_p, ca_bv, s2, MN, DD, DD, false);           // V
        head_attn_kernel<<<(MN * 8 + 255) / 256, 256, 0, stream>>>(s0, s1, s2, s3, MN);
        packA(s3, pA, MN, DD);
        gemm(pA, wo_p, ca_bo, s1, MN, DD, DD, false);           // O proj
        resln_kernel<<<MN, 256, 0, stream>>>(fx, s1, ca_ln1g, ca_ln1b, s0);   // x
        packA(s0, pA, MN, DD);
        gemm(pA, w1_p, ca_b1, sH, MN, 2 * DD, DD, true);        // FFN up + ReLU
        packA(sH, pH, MN, 2 * DD);
        gemm(pH, w2_p, ca_b2, s1, MN, DD, 2 * DD, false);       // FFN down
        resln_kernel<<<MN, 256, 0, stream>>>(s0, s1, ca_ln2g, ca_ln2b, e);    // e
    };

    attend(f1, f2, e1);
    attend(f2, f1, e2);

    // ---- Metric similarity
    msym_kernel<<<(DD * DD) / 256, 256, 0, stream>>>(ms_met, msym);
    packB(msym, msym_p, DD, DD);
    packA(e1, pA, MN, DD);
    packA(e2, pB, MN, DD);
    gemm(pA, mswt_p, ms_bt, s0, MN, DD, DD, false);             // t1
    gemm(pB, mswt_p, ms_bt, s1, MN, DD, DD, false);             // t2
    packA(s0, pA, MN, DD);
    packA(s1, pB, MN, DD);
    gemm(pA, msym_p, nullptr, s2, MN, DD, DD, false);           // t1M
    gemm(pB, msym_p, nullptr, s3, MN, DD, DD, false);           // t2M
    rowq_kernel<<<MN / 256, 256, 0, stream>>>(s2, s0, q1w, MN);
    rowq_kernel<<<MN / 256, 256, 0, stream>>>(s3, s1, q2w, MN);
    cross_sim_kernel<<<dim3(BB, (SS + 255) / 256), 256, 0, stream>>>(s2, s1, q1w, q2w, flat);

    // ---- Attention pooling
    zero_kernel<<<(BB * FP + 255) / 256, 256, 0, stream>>>(accu, BB * FP);
    pool_attn_kernel<<<dim3(BB, (SS + 31) / 32), 256, 0, stream>>>(
        flat, ap_wq, ap_bq, ap_wk, ap_bk, ap_wv, ap_bv, ap_pos, accu);
    pool_final_kernel<<<BB, 64, 0, stream>>>(accu, ap_wo, ap_bo, ap_lng, ap_lnb,
                                             (float*)d_out);
}
